// FusionBlock_78975858639181
// MI455X (gfx1250) — matrix operand order repaired and checked by reading, not compiled, not run on hardware
//
#include <hip/hip_runtime.h>
#include <hip/hip_bf16.h>

// Problem constants (from reference setup_inputs)
#define MM 2048   // tokens
#define NN 256    // graph nodes
#define DD 256    // d2
#define LL 64     // query length
#define K2 512    // 2*d2

typedef __attribute__((ext_vector_type(16))) __bf16 v16bf;
typedef __attribute__((ext_vector_type(8)))  float  v8f;

__device__ __forceinline__ __bf16 f2bf(float f) {
    union { float f; unsigned u; } x; x.f = f;
    unsigned r = x.u + 0x7FFFu + ((x.u >> 16) & 1u);   // round-to-nearest-even
    union { unsigned short s; __bf16 b; } y; y.s = (unsigned short)(r >> 16);
    return y.b;
}

__device__ __forceinline__ float sigmoidf(float x) { return 1.0f / (1.0f + expf(-x)); }

union FragU { v16bf v; uint4 q[2]; };

// A fragment (16x32 bf16, row-major A, lda in elements, kb multiple of 32).
// Lane l: row = mb + (l&15), h = l>>4; VGPR pairs hold K = {h*8..h*8+7, 16+h*8..}.
__device__ __forceinline__ v16bf load_afrag(const __bf16* A, int lda, int mb, int kb, int lane) {
    int m = mb + (lane & 15);
    int h = lane >> 4;
    const uint4* p = (const uint4*)(A + (size_t)m * lda + kb);
    FragU u;
    u.q[0] = p[h];        // K = h*8 .. h*8+7
    u.q[1] = p[2 + h];    // K = 16+h*8 .. 16+h*8+7
    return u.v;
}

// B fragment (32x16 bf16) from B^T stored row-major (row j, K contiguous).
// Lane l: col = jb + (l&15), h = l>>4; VGPR pairs hold K = {h*8..h*8+7, 16+h*8..}.
__device__ __forceinline__ v16bf load_bfrag(const __bf16* Bt, int ldb, int jb, int kb, int lane) {
    int j = jb + (lane & 15);
    int h = lane >> 4;
    const uint4* p = (const uint4*)(Bt + (size_t)j * ldb + kb);
    FragU u;
    u.q[0] = p[h];
    u.q[1] = p[2 + h];
    return u.v;
}

__device__ __forceinline__ v8f wmma_bf16(v16bf a, v16bf b, v8f c) {
    // (neg_a, A, neg_b, B, c_mod, C, reuse_a, reuse_b)
    return __builtin_amdgcn_wmma_f32_16x16x32_bf16(false, a, false, b, (short)0, c, false, false);
}

// ---------------- elementwise / reduction kernels ----------------

__global__ void cvt_kernel(const float* __restrict__ in, __bf16* __restrict__ out, int n) {
    int i = blockIdx.x * blockDim.x + threadIdx.x;
    for (; i < n; i += gridDim.x * blockDim.x) out[i] = f2bf(in[i]);
}

// ent[n, 0:256] = (bin_M.T @ ctx)[n]/M ; ent[n, 256:512] = max_m (bin*ctx)
__global__ void pool_kernel(const float* __restrict__ ctx, const float* __restrict__ binM,
                            float* __restrict__ ent) {
    int n = blockIdx.x, d = threadIdx.x;
    float s = 0.f, mx = -INFINITY;
    for (int m = 0; m < MM; ++m) {
        float bv = binM[(size_t)m * NN + n];   // uniform across block -> broadcast
        float xv = ctx[(size_t)m * DD + d];
        float p = bv * xv;
        s += p;
        mx = fmaxf(mx, p);
    }
    ent[(size_t)n * K2 + d]      = s * (1.0f / (float)MM);
    ent[(size_t)n * K2 + DD + d] = mx;
}

__global__ void qmean_kernel(const float* __restrict__ q, float* __restrict__ qm) {
    int d = threadIdx.x;
    float s = 0.f;
    for (int l = 0; l < LL; ++l) s += q[(size_t)l * DD + d];
    qm[d] = s * (1.0f / (float)LL);
}

// vq[k] = sum_d V[d,k] * qm[d]   (V is d2 x 2*d2)
__global__ void vq_kernel(const float* __restrict__ V, const float* __restrict__ qm,
                          float* __restrict__ vq) {
    int k = blockIdx.x * 256 + threadIdx.x;   // 0..511
    float s = 0.f;
    for (int d = 0; d < DD; ++d) s += V[(size_t)d * K2 + k] * qm[d];
    vq[k] = s;
}

// gamma[n] = dot(ent[n], vq)/sqrt(d2); entg = sigmoid(gamma) * ent  (bf16)
__global__ void gate_entg_kernel(const float* __restrict__ ent, const float* __restrict__ vq,
                                 __bf16* __restrict__ entg) {
    __shared__ float red[256];
    int n = blockIdx.x, t = threadIdx.x;
    const float* er = ent + (size_t)n * K2;
    red[t] = er[t] * vq[t] + er[DD + t] * vq[DD + t];
    __syncthreads();
    for (int s = 128; s > 0; s >>= 1) {
        if (t < s) red[t] += red[t + s];
        __syncthreads();
    }
    float gate = sigmoidf(red[0] * (1.0f / 16.0f));   // sqrt(256) == 16
    entg[(size_t)n * K2 + t]      = f2bf(gate * er[t]);
    entg[(size_t)n * K2 + DD + t] = f2bf(gate * er[DD + t]);
}

// si[i] = hidden[i,:].W[:256], sj[i] = hidden[i,:].W[256:]  (hidden stored transposed)
__global__ void sisj_kernel(const float* __restrict__ hidT, const float* __restrict__ W,
                            float* __restrict__ si, float* __restrict__ sj) {
    int i = threadIdx.x;
    float a = 0.f, bz = 0.f;
    for (int d = 0; d < DD; ++d) {
        float h = hidT[(size_t)d * NN + i];
        a  += h * W[d];
        bz += h * W[DD + d];
    }
    si[i] = a; sj[i] = bz;
}

// One block per row j of betas; writes P[i][j] = A[i][j]*alphas[j][i] in bf16
__global__ void softmaxP_kernel(const float* __restrict__ si, const float* __restrict__ sj,
                                const int* __restrict__ adj, __bf16* __restrict__ P) {
    __shared__ float red[256];
    int j = blockIdx.x, i = threadIdx.x;
    float s = si[j] + sj[i];
    s = (s > 0.f) ? s : 0.01f * s;                     // leaky_relu
    float Aji = (adj[(size_t)j * NN + i] > 0) ? 1.f : 0.f;
    float e = expf(Aji * s);                           // non-edges contribute exp(0)=1
    red[i] = e;
    __syncthreads();
    for (int st = 128; st > 0; st >>= 1) {
        if (i < st) red[i] += red[i + st];
        __syncthreads();
    }
    float alpha = e / red[0];
    float Aij = (adj[(size_t)i * NN + j] > 0) ? 1.f : 0.f;
    P[(size_t)i * NN + j] = f2bf(Aij * alpha);
}

// ---------------- WMMA GEMM kernels (one 16x16 tile per wave32) ----------------

// hidden = entg @ U^T + b; store hidden transposed (f32 + bf16), K=512
__global__ void hidden_gemm_kernel(const __bf16* __restrict__ entg, const __bf16* __restrict__ U_bf,
                                   const float* __restrict__ b,
                                   float* __restrict__ hidT_f, __bf16* __restrict__ hidT_bf) {
    int wid = threadIdx.x >> 5, lane = threadIdx.x & 31;
    int tile = blockIdx.x * 8 + wid;     // 256 tiles
    int mb = (tile >> 4) * 16;           // node row
    int jb = (tile & 15) * 16;           // hidden col
    v8f c = {};
    for (int kb = 0; kb < K2; kb += 32) {
        v16bf a  = load_afrag(entg, K2, mb, kb, lane);
        v16bf bb = load_bfrag(U_bf, K2, jb, kb, lane);
        c = wmma_bf16(a, bb, c);
    }
    int col = jb + (lane & 15);
    int row0 = mb + (lane >> 4) * 8;
    float bj = b[col];
#pragma unroll
    for (int r = 0; r < 8; ++r) {
        float v = c[r] + bj;
        hidT_f[(size_t)col * NN + row0 + r]  = v;        // contiguous per lane
        hidT_bf[(size_t)col * NN + row0 + r] = f2bf(v);
    }
}

// E_t = relu(P @ hidden); store transposed bf16, K=256
__global__ void et_gemm_kernel(const __bf16* __restrict__ P, const __bf16* __restrict__ hidT_bf,
                               __bf16* __restrict__ EtT) {
    int wid = threadIdx.x >> 5, lane = threadIdx.x & 31;
    int tile = blockIdx.x * 8 + wid;     // 256 tiles
    int mb = (tile >> 4) * 16;
    int jb = (tile & 15) * 16;
    v8f c = {};
    for (int kb = 0; kb < NN; kb += 32) {
        v16bf a  = load_afrag(P, NN, mb, kb, lane);
        v16bf bb = load_bfrag(hidT_bf, NN, jb, kb, lane);
        c = wmma_bf16(a, bb, c);
    }
    int col = jb + (lane & 15);
    int row0 = mb + (lane >> 4) * 8;
#pragma unroll
    for (int r = 0; r < 8; ++r)
        EtT[(size_t)col * NN + row0 + r] = f2bf(fmaxf(c[r], 0.f));
}

// emb_info = bin_M @ E_t; row-major bf16 output, K=256
__global__ void embinfo_gemm_kernel(const __bf16* __restrict__ binM_bf, const __bf16* __restrict__ EtT,
                                    __bf16* __restrict__ emb) {
    int wid = threadIdx.x >> 5, lane = threadIdx.x & 31;
    int tile = blockIdx.x * 8 + wid;     // 2048 tiles
    int mb = (tile >> 4) * 16;
    int jb = (tile & 15) * 16;
    v8f c = {};
    for (int kb = 0; kb < NN; kb += 32) {
        v16bf a  = load_afrag(binM_bf, NN, mb, kb, lane);
        v16bf bb = load_bfrag(EtT, NN, jb, kb, lane);
        c = wmma_bf16(a, bb, c);
    }
    int col = jb + (lane & 15);
    int row0 = mb + (lane >> 4) * 8;
#pragma unroll
    for (int r = 0; r < 8; ++r)
        emb[(size_t)(row0 + r) * DD + col] = f2bf(c[r]);
}

// gates = [ctx|emb] @ W_ih^T + (b_ih+b_hh); fused LSTM epilogue.
// f-gate and W_hh are dead because h0 = c0 = 0.
__global__ void gates_lstm_kernel(const __bf16* __restrict__ ctx_bf, const __bf16* __restrict__ emb_bf,
                                  const __bf16* __restrict__ Wih_bf,
                                  const float* __restrict__ b_ih, const float* __restrict__ b_hh,
                                  float* __restrict__ out) {
    int wid = threadIdx.x >> 5, lane = threadIdx.x & 31;
    int tile = blockIdx.x * 8 + wid;     // 2048 tiles
    int mb = (tile >> 4) * 16;           // token rows
    int db = (tile & 15) * 16;           // output feature cols
    v8f ci = {}, cg = {}, co = {};
    for (int kb = 0; kb < K2; kb += 32) {
        v16bf a = (kb < DD) ? load_afrag(ctx_bf, DD, mb, kb, lane)
                            : load_afrag(emb_bf, DD, mb, kb - DD, lane);
        v16bf bi = load_bfrag(Wih_bf, K2, db, kb, lane);            // i-gate rows [0,256)
        v16bf bg = load_bfrag(Wih_bf, K2, 2 * DD + db, kb, lane);   // g-gate rows [512,768)
        v16bf bo = load_bfrag(Wih_bf, K2, 3 * DD + db, kb, lane);   // o-gate rows [768,1024)
        ci = wmma_bf16(a, bi, ci);
        cg = wmma_bf16(a, bg, cg);
        co = wmma_bf16(a, bo, co);
    }
    int d = db + (lane & 15);
    int row0 = mb + (lane >> 4) * 8;
    float Bi = b_ih[d] + b_hh[d];
    float Bg = b_ih[2 * DD + d] + b_hh[2 * DD + d];
    float Bo = b_ih[3 * DD + d] + b_hh[3 * DD + d];
#pragma unroll
    for (int r = 0; r < 8; ++r) {
        float cc = sigmoidf(ci[r] + Bi) * tanhf(cg[r] + Bg);
        out[(size_t)(row0 + r) * DD + d] = sigmoidf(co[r] + Bo) * tanhf(cc);
    }
}

// ---------------- host side ----------------

extern "C" void kernel_launch(void* const* d_in, const int* in_sizes, int n_in,
                              void* d_out, int out_size, void* d_ws, size_t ws_size,
                              hipStream_t stream) {
    (void)in_sizes; (void)n_in; (void)out_size; (void)ws_size;
    const float* ctx   = (const float*)d_in[0];
    const float* query = (const float*)d_in[1];
    const float* binM  = (const float*)d_in[2];
    const float* V     = (const float*)d_in[3];
    const float* U     = (const float*)d_in[4];
    const float* b     = (const float*)d_in[5];
    const float* W     = (const float*)d_in[6];
    const float* W_ih  = (const float*)d_in[7];
    /* d_in[8] = W_hh: unused, h0 == 0 */
    const float* b_ih  = (const float*)d_in[9];
    const float* b_hh  = (const float*)d_in[10];
    const int*   adj   = (const int*)d_in[11];
    float* out = (float*)d_out;

    char* w = (char*)d_ws;
    size_t off = 0;
    auto alloc = [&](size_t bytes) -> void* {
        void* p = w + off;
        off += (bytes + 255) & ~(size_t)255;
        return p;
    };
    __bf16* ctx_bf  = (__bf16*)alloc((size_t)MM * DD * 2);
    __bf16* binM_bf = (__bf16*)alloc((size_t)MM * NN * 2);
    __bf16* U_bf    = (__bf16*)alloc((size_t)DD * K2 * 2);
    __bf16* Wih_bf  = (__bf16*)alloc((size_t)4 * DD * K2 * 2);
    float*  ent     = (float*) alloc((size_t)NN * K2 * 4);
    float*  qm      = (float*) alloc((size_t)DD * 4);
    float*  vq      = (float*) alloc((size_t)K2 * 4);
    __bf16* entg    = (__bf16*)alloc((size_t)NN * K2 * 2);
    float*  hidT_f  = (float*) alloc((size_t)NN * DD * 4);
    __bf16* hidT_bf = (__bf16*)alloc((size_t)NN * DD * 2);
    float*  si      = (float*) alloc((size_t)NN * 4);
    float*  sj      = (float*) alloc((size_t)NN * 4);
    __bf16* P_bf    = (__bf16*)alloc((size_t)NN * NN * 2);
    __bf16* EtT_bf  = (__bf16*)alloc((size_t)NN * DD * 2);
    __bf16* emb_bf  = (__bf16*)alloc((size_t)MM * DD * 2);

    cvt_kernel<<<512, 256, 0, stream>>>(ctx,  ctx_bf,  MM * DD);
    cvt_kernel<<<512, 256, 0, stream>>>(binM, binM_bf, MM * NN);
    cvt_kernel<<<256, 256, 0, stream>>>(U,    U_bf,    DD * K2);
    cvt_kernel<<<512, 256, 0, stream>>>(W_ih, Wih_bf,  4 * DD * K2);

    pool_kernel<<<NN, DD, 0, stream>>>(ctx, binM, ent);
    qmean_kernel<<<1, DD, 0, stream>>>(query, qm);
    vq_kernel<<<2, 256, 0, stream>>>(V, qm, vq);
    gate_entg_kernel<<<NN, 256, 0, stream>>>(ent, vq, entg);

    hidden_gemm_kernel<<<32, 256, 0, stream>>>(entg, U_bf, b, hidT_f, hidT_bf);
    sisj_kernel<<<1, NN, 0, stream>>>(hidT_f, W, si, sj);
    softmaxP_kernel<<<NN, NN, 0, stream>>>(si, sj, adj, P_bf);
    et_gemm_kernel<<<32, 256, 0, stream>>>(P_bf, hidT_bf, EtT_bf);
    embinfo_gemm_kernel<<<256, 256, 0, stream>>>(binM_bf, EtT_bf, emb_bf);
    gates_lstm_kernel<<<256, 256, 0, stream>>>(ctx_bf, emb_bf, Wih_bf, b_ih, b_hh, out);
}
